// RSSM_67250597921080
// MI455X (gfx1250) — compile-verified, matrix operand order checked by reading, not run on hardware
//
#include <hip/hip_runtime.h>
#include <hip/hip_bf16.h>
#include <math.h>

// ---------------------------------------------------------------------------
// RSSM rollout for MI455X (gfx1250, wave32, WMMA 16x16x32 bf16).
// B=1024, T=128, D=256, S=32, H=256, OBS=64, ACT=8. Out = [B,T,416] f32.
// 64 blocks x 256 threads; each block owns 16 batch rows for all 128 steps.
// ---------------------------------------------------------------------------

typedef __attribute__((ext_vector_type(16))) __bf16 v16bf;
typedef __attribute__((ext_vector_type(8)))  __bf16 v8bf;
typedef __attribute__((ext_vector_type(8)))  float  v8f;

#define Bsz 1024
#define Tsz 128
#define OC  416

// padded LDS strides (avoid power-of-2 bank aliasing)
#define SF 260   // f32 row stride (elements) for 16x256 buffers
#define SB 264   // bf16 row stride for 16x256 buffers
#define SX 72    // bf16 row stride for 16x64 buffers

// ---------------------------------------------------------------------------
// Fast transcendentals on gfx1250 hardware units (v_exp_f32 / v_rcp_f32 /
// v_tanh_f32). Branchless; keeps live ranges tight around WMMA accumulators.
// ---------------------------------------------------------------------------
__device__ inline float fexp(float x) {
#if __has_builtin(__builtin_amdgcn_exp2f)
    return __builtin_amdgcn_exp2f(x * 1.4426950408889634f);
#else
    return __expf(x);
#endif
}
__device__ inline float frcp(float x) {
#if __has_builtin(__builtin_amdgcn_rcpf)
    return __builtin_amdgcn_rcpf(x);
#else
    return 1.f / x;
#endif
}
__device__ inline float ftanh(float x) {
#if __has_builtin(__builtin_amdgcn_tanhf)
    return __builtin_amdgcn_tanhf(x);
#else
    float e = fexp(-2.f * x);
    return (1.f - e) * frcp(1.f + e);
#endif
}
__device__ inline float sigm(float x) { return frcp(1.f + fexp(-x)); }
__device__ inline float elu(float x)  { float e = fexp(x) - 1.f; return x > 0.f ? x : e; }
__device__ inline float expclip(float x) { return fexp(fminf(fmaxf(x, -7.f), 5.f)); }

// ---------------------------------------------------------------------------
// A-fragment load from LDS row-major [16][K] bf16.
// Lane layout (16-bit A, 16x32): lane = m + 16*h; elements j0..7 -> k = 32kt+8h+j,
// j8..15 -> k = 32kt+16+8h+(j-8). Two 16B ds loads per fragment.
// ---------------------------------------------------------------------------
__device__ inline v16bf ldsA(const __bf16* row, int kt, int h) {
    const __bf16* p = row + kt * 32 + h * 8;
    v8bf lo = *(const v8bf*)(p);
    v8bf hi = *(const v8bf*)(p + 16);
    return __builtin_shufflevector(lo, hi, 0,1,2,3,4,5,6,7,8,9,10,11,12,13,14,15);
}

__device__ inline v8f wmma_bf16(v16bf a, v16bf b, v8f c) {
    return __builtin_amdgcn_wmma_f32_16x16x32_bf16(false, a, false, b,
                                                   (short)0, c, false, false);
}

// ---------------------------------------------------------------------------
// Single-tile GEMM, ping-pong double buffered (no rotation movs; 2 B-loads
// in flight; unroll 1 keeps register pressure bounded). nkt must be even.
// ---------------------------------------------------------------------------
__device__ inline v8f gemm1(v8f c, const __bf16* A, int astride,
                            const __bf16* Wp, int WKT, int nt,
                            int kt0, int nkt, int lane) {
    const int m = lane & 15, h = lane >> 4;
    const __bf16* arow = A + m * astride;
    const __bf16* bp = Wp + ((size_t)((nt * WKT) + kt0) * 32 + lane) * 16;
    v16bf a0 = ldsA(arow, 0, h);
    v16bf b0 = *(const v16bf*)(bp);
    v16bf a1 = ldsA(arow, 1, h);
    v16bf b1 = *(const v16bf*)(bp + 512);
#pragma unroll 1
    for (int kt = 0; kt < nkt - 2; kt += 2) {
        c  = wmma_bf16(a0, b0, c);
        a0 = ldsA(arow, kt + 2, h);
        b0 = *(const v16bf*)(bp + (size_t)(kt + 2) * 512);
        c  = wmma_bf16(a1, b1, c);
        a1 = ldsA(arow, kt + 3, h);
        b1 = *(const v16bf*)(bp + (size_t)(kt + 3) * 512);
    }
    c = wmma_bf16(a0, b0, c);
    return wmma_bf16(a1, b1, c);
}

// ---------------------------------------------------------------------------
// Dual-tile GEMM: two n-tiles share every A fragment (half the LDS traffic)
// and interleave two independent accumulator chains (hides WMMA RAW latency).
// nkt must be even. Live set: 6 fragments + 2 accs -> no spills at unroll 1.
// ---------------------------------------------------------------------------
__device__ inline void gemm2(v8f& c0, v8f& c1, const __bf16* A, int astride,
                             const __bf16* Wp, int WKT, int nt0, int nt1,
                             int kt0, int nkt, int lane) {
    const int m = lane & 15, h = lane >> 4;
    const __bf16* arow = A + m * astride;
    const __bf16* bp0 = Wp + ((size_t)((nt0 * WKT) + kt0) * 32 + lane) * 16;
    const __bf16* bp1 = Wp + ((size_t)((nt1 * WKT) + kt0) * 32 + lane) * 16;
    v16bf a0  = ldsA(arow, 0, h);
    v16bf b00 = *(const v16bf*)(bp0);
    v16bf b10 = *(const v16bf*)(bp1);
    v16bf a1  = ldsA(arow, 1, h);
    v16bf b01 = *(const v16bf*)(bp0 + 512);
    v16bf b11 = *(const v16bf*)(bp1 + 512);
#pragma unroll 1
    for (int kt = 0; kt < nkt - 2; kt += 2) {
        c0  = wmma_bf16(a0, b00, c0);
        c1  = wmma_bf16(a0, b10, c1);
        a0  = ldsA(arow, kt + 2, h);
        b00 = *(const v16bf*)(bp0 + (size_t)(kt + 2) * 512);
        b10 = *(const v16bf*)(bp1 + (size_t)(kt + 2) * 512);
        c0  = wmma_bf16(a1, b01, c0);
        c1  = wmma_bf16(a1, b11, c1);
        a1  = ldsA(arow, kt + 3, h);
        b01 = *(const v16bf*)(bp0 + (size_t)(kt + 3) * 512);
        b11 = *(const v16bf*)(bp1 + (size_t)(kt + 3) * 512);
    }
    c0 = wmma_bf16(a0, b00, c0);
    c1 = wmma_bf16(a0, b10, c1);
    c0 = wmma_bf16(a1, b01, c0);
    c1 = wmma_bf16(a1, b11, c1);
}

// D layout: lane = n_local + 16*h holds M = v + 8*h for VGPR v.
__device__ inline void store_elu_bf(v8f acc, const float* bias, int col, int h,
                                    __bf16* dst) {
    const float b = bias[col];
#pragma unroll
    for (int v = 0; v < 8; ++v) {
        float x = elu(acc[v] + b);
        dst[(v + 8 * h) * SB + col] = (__bf16)x;
    }
}

// ---------------------------------------------------------------------------
// Weight pre-pack: W[N,K] row-major f32 -> bf16 B-fragments, layout
// dst[((nt*KT + kt)*32 + lane)*16 + j], lane = n_local + 16*h, K interleave as A.
// ---------------------------------------------------------------------------
__global__ void pack_kernel(const float* __restrict__ W, __bf16* __restrict__ dst,
                            int K, int KT, int total) {
    int p = blockIdx.x * 256 + threadIdx.x;
    if (p >= total) return;
    int j    = p & 15;
    int lane = (p >> 4) & 31;
    int kt   = (p >> 9) % KT;
    int nt   = (p >> 9) / KT;
    int n = nt * 16 + (lane & 15);
    int h = lane >> 4;
    int kin = (j < 8) ? (h * 8 + j) : (16 + h * 8 + (j - 8));
    int k = kt * 32 + kin;
    float v = (k < K) ? W[(size_t)n * K + k] : 0.f;  // zero-pads W_ih K=40 -> 64
    dst[p] = (__bf16)v;
}

// ---------------------------------------------------------------------------
// Main persistent-tile kernel. One block = 16 batch rows for all T steps.
// ---------------------------------------------------------------------------
__global__ __launch_bounds__(256)
void rssm_kernel(const float* __restrict__ obs, const float* __restrict__ act,
                 const float* __restrict__ noise,
                 const float* __restrict__ b_e1, const float* __restrict__ b_e2,
                 const float* __restrict__ b_ih, const float* __restrict__ b_hh,
                 const float* __restrict__ b_p1, const float* __restrict__ b_p2,
                 const float* __restrict__ b_q1, const float* __restrict__ b_q2,
                 const __bf16* __restrict__ Pe1, const __bf16* __restrict__ Pe2,
                 const __bf16* __restrict__ Pih, const __bf16* __restrict__ Phh,
                 const __bf16* __restrict__ Pp1, const __bf16* __restrict__ Pp2,
                 const __bf16* __restrict__ Pq1, const __bf16* __restrict__ Pq2,
                 float* __restrict__ out) {
    __shared__ __align__(16) float  h_f32[16 * SF];
    __shared__ __align__(16) float  r_f32[16 * SF];
    __shared__ __align__(16) float  u_f32[16 * SF];   // reused as h_new
    __shared__ __align__(16) float  ghn_f32[16 * SF];
    __shared__ __align__(16) __bf16 h_bf[16 * SB];
    __shared__ __align__(16) __bf16 emb_bf[16 * SB];
    __shared__ __align__(16) __bf16 tmp_bf[16 * SB];  // P1 out, then Q1 out
    __shared__ __align__(16) __bf16 e1_bf[16 * SB];
    __shared__ __align__(16) __bf16 obs_bf[16 * SX];
    __shared__ __align__(16) __bf16 x_bf[16 * SX];    // [z(32) | a(8) | pad(24)]
    __shared__ __align__(16) float  qm_f32[16 * 32];
    __shared__ __align__(16) float  qs_f32[16 * 32];

    const int tid  = threadIdx.x;
    const int lane = tid & 31;
    const int wave = tid >> 5;
    const int nl   = lane & 15;
    const int hh   = lane >> 4;
    const int b0   = blockIdx.x * 16;

    // init: h = 0, x padding = 0
    for (int e = tid; e < 16 * 256; e += 256) {
        int m = e >> 8, c = e & 255;
        h_f32[m * SF + c] = 0.f;
        h_bf[m * SB + c]  = (__bf16)0.f;
    }
    for (int e = tid; e < 16 * 24; e += 256) {
        int m = e / 24, c = e % 24;
        x_bf[m * SX + 40 + c] = (__bf16)0.f;
    }
    __syncthreads();

    for (int t = 0; t < Tsz; ++t) {
        const int nt0 = wave * 2, nt1 = wave * 2 + 1;

        // ---- Phase A: emit h_old, stage obs & act -------------------------
        for (int e = tid; e < 4096; e += 256) {
            int m = e >> 8, c = e & 255;
            out[((size_t)(b0 + m) * Tsz + t) * OC + c] = h_f32[m * SF + c];
        }
        for (int e = tid; e < 1024; e += 256) {
            int m = e >> 6, k = e & 63;
            obs_bf[m * SX + k] = (__bf16)obs[((size_t)(b0 + m) * Tsz + t) * 64 + k];
        }
        if (tid < 128) {
            int m = tid >> 3, j = tid & 7;
            x_bf[m * SX + 32 + j] = (__bf16)act[((size_t)(b0 + m) * Tsz + t) * 8 + j];
        }
        __syncthreads();

        // ---- Phase B: P1 (tmp = ELU(h@Wp1^T+b)), E1 (e1 = ELU(o@We1^T+b)) --
        {
            v8f c0 = {}, c1 = {};
            gemm2(c0, c1, h_bf, SB, Pp1, 8, nt0, nt1, 0, 8, lane);
            store_elu_bf(c0, b_p1, nt0 * 16 + nl, hh, tmp_bf);
            store_elu_bf(c1, b_p1, nt1 * 16 + nl, hh, tmp_bf);
            v8f d0 = {}, d1 = {};
            gemm2(d0, d1, obs_bf, SX, Pe1, 2, nt0, nt1, 0, 2, lane);
            store_elu_bf(d0, b_e1, nt0 * 16 + nl, hh, e1_bf);
            store_elu_bf(d1, b_e1, nt1 * 16 + nl, hh, e1_bf);
        }
        __syncthreads();

        // ---- Phase C: E2 (all waves) + P2 stats (waves 0-3) ---------------
        {
            v8f c0 = {}, c1 = {};
            gemm2(c0, c1, e1_bf, SB, Pe2, 8, nt0, nt1, 0, 8, lane);
            store_elu_bf(c0, b_e2, nt0 * 16 + nl, hh, emb_bf);
            store_elu_bf(c1, b_e2, nt1 * 16 + nl, hh, emb_bf);
        }
        if (wave < 4) {
            v8f acc = {};
            acc = gemm1(acc, tmp_bf, SB, Pp2, 8, wave, 0, 8, lane);
            int col = wave * 16 + nl;
            float bias = b_p2[col];
#pragma unroll
            for (int v = 0; v < 8; ++v) {
                int m = v + 8 * hh;
                size_t ob = ((size_t)(b0 + m) * Tsz + t) * OC;
                float x = acc[v] + bias;
                if (col < 32) out[ob + 288 + col] = x;                 // pm
                else          out[ob + 320 + (col - 32)] = expclip(x); // ps
            }
        }
        __syncthreads();

        // ---- Phase D: Q1 (K=512 = [h | emb]) -> tmp_bf --------------------
        {
            v8f c0 = {}, c1 = {};
            gemm2(c0, c1, h_bf,   SB, Pq1, 16, nt0, nt1, 0, 8, lane);
            gemm2(c0, c1, emb_bf, SB, Pq1, 16, nt0, nt1, 8, 8, lane);
            store_elu_bf(c0, b_q1, nt0 * 16 + nl, hh, tmp_bf);
            store_elu_bf(c1, b_q1, nt1 * 16 + nl, hh, tmp_bf);
        }
        __syncthreads();

        // ---- Phase E: gh_n pre-pass (all waves) + Q2 stats (waves 0-3) ----
        {
            v8f c0 = {}, c1 = {};
            gemm2(c0, c1, h_bf, SB, Phh, 8, 32 + nt0, 32 + nt1, 0, 8, lane);
#pragma unroll
            for (int v = 0; v < 8; ++v) {
                int m = v + 8 * hh;
                int col0 = nt0 * 16 + nl, col1 = nt1 * 16 + nl;
                ghn_f32[m * SF + col0] = c0[v] + b_hh[512 + col0];
                ghn_f32[m * SF + col1] = c1[v] + b_hh[512 + col1];
            }
        }
        if (wave < 4) {
            v8f acc = {};
            acc = gemm1(acc, tmp_bf, SB, Pq2, 8, wave, 0, 8, lane);
            int col = wave * 16 + nl;
            float bias = b_q2[col];
#pragma unroll
            for (int v = 0; v < 8; ++v) {
                int m = v + 8 * hh;
                size_t ob = ((size_t)(b0 + m) * Tsz + t) * OC;
                float x = acc[v] + bias;
                if (col < 32) { qm_f32[m * 32 + col] = x; out[ob + 352 + col] = x; }
                else {
                    float s = expclip(x);
                    qs_f32[m * 32 + (col - 32)] = s;
                    out[ob + 384 + (col - 32)] = s;
                }
            }
        }
        __syncthreads();

        // ---- z = qm + qs * eps ; feed GRU input ---------------------------
        for (int e = tid; e < 512; e += 256) {
            int m = e >> 5, s = e & 31;
            float eps = noise[((size_t)t * Bsz + (b0 + m)) * 32 + s];
            float z = qm_f32[m * 32 + s] + qs_f32[m * 32 + s] * eps;
            out[((size_t)(b0 + m) * Tsz + t) * OC + 256 + s] = z;
            x_bf[m * SX + s] = (__bf16)z;
        }
        __syncthreads();

        // ---- G1: r,u gates (tiles 0..31): acc = x@Wih^T + h@Whh^T ---------
        for (int p = 0; p < 2; ++p) {
            int tt0 = wave * 4 + p * 2, tt1 = tt0 + 1;
            v8f c0 = {}, c1 = {};
            gemm2(c0, c1, x_bf, SX, Pih, 2, tt0, tt1, 0, 2, lane);
            gemm2(c0, c1, h_bf, SB, Phh, 8, tt0, tt1, 0, 8, lane);
            int col0 = tt0 * 16 + nl, col1 = tt1 * 16 + nl;
            float bi0 = b_ih[col0] + b_hh[col0];
            float bi1 = b_ih[col1] + b_hh[col1];
#pragma unroll
            for (int v = 0; v < 8; ++v) {
                int m = v + 8 * hh;
                float g0 = sigm(c0[v] + bi0);
                float g1 = sigm(c1[v] + bi1);
                if (col0 < 256) r_f32[m * SF + col0] = g0;
                else            u_f32[m * SF + (col0 - 256)] = g0;
                if (col1 < 256) r_f32[m * SF + col1] = g1;
                else            u_f32[m * SF + (col1 - 256)] = g1;
            }
        }
        __syncthreads();

        // ---- G2: n gate + h update (h_new stored in-place in u_f32) -------
        {
            v8f c0 = {}, c1 = {};
            gemm2(c0, c1, x_bf, SX, Pih, 2, 32 + nt0, 32 + nt1, 0, 2, lane);
            int col0 = nt0 * 16 + nl, col1 = nt1 * 16 + nl;
            float bi0 = b_ih[512 + col0], bi1 = b_ih[512 + col1];
#pragma unroll
            for (int v = 0; v < 8; ++v) {
                int m = v + 8 * hh;
                float n0 = ftanh(c0[v] + bi0 +
                                 r_f32[m * SF + col0] * ghn_f32[m * SF + col0]);
                float u0 = u_f32[m * SF + col0];
                u_f32[m * SF + col0] = (1.f - u0) * n0 + u0 * h_f32[m * SF + col0];
                float n1 = ftanh(c1[v] + bi1 +
                                 r_f32[m * SF + col1] * ghn_f32[m * SF + col1]);
                float u1 = u_f32[m * SF + col1];
                u_f32[m * SF + col1] = (1.f - u1) * n1 + u1 * h_f32[m * SF + col1];
            }
        }
        __syncthreads();

        // ---- H: commit recurrent state ------------------------------------
        for (int e = tid; e < 4096; e += 256) {
            int m = e >> 8, c = e & 255;
            float f = u_f32[m * SF + c];
            h_f32[m * SF + c] = f;
            h_bf[m * SB + c]  = (__bf16)f;
        }
        __syncthreads();
    }
}

// ---------------------------------------------------------------------------
extern "C" void kernel_launch(void* const* d_in, const int* in_sizes, int n_in,
                              void* d_out, int out_size, void* d_ws, size_t ws_size,
                              hipStream_t stream) {
    const float* obs  = (const float*)d_in[0];
    const float* actp = (const float*)d_in[1];
    const float* noi  = (const float*)d_in[2];
    const float* W_e1 = (const float*)d_in[3];
    const float* b_e1 = (const float*)d_in[4];
    const float* W_e2 = (const float*)d_in[5];
    const float* b_e2 = (const float*)d_in[6];
    const float* W_ih = (const float*)d_in[7];
    const float* W_hh = (const float*)d_in[8];
    const float* b_ih = (const float*)d_in[9];
    const float* b_hh = (const float*)d_in[10];
    const float* W_p1 = (const float*)d_in[11];
    const float* b_p1 = (const float*)d_in[12];
    const float* W_p2 = (const float*)d_in[13];
    const float* b_p2 = (const float*)d_in[14];
    const float* W_q1 = (const float*)d_in[15];
    const float* b_q1 = (const float*)d_in[16];
    const float* W_q2 = (const float*)d_in[17];
    const float* b_q2 = (const float*)d_in[18];
    float* out = (float*)d_out;

    __bf16* ws = (__bf16*)d_ws;
    __bf16* Pe1 = ws;                 // N=256 K=64  KT=2  -> 16384
    __bf16* Pe2 = Pe1 + 16384;        // N=256 K=256 KT=8  -> 65536
    __bf16* Pih = Pe2 + 65536;        // N=768 K=40->64 KT=2 -> 49152
    __bf16* Phh = Pih + 49152;        // N=768 K=256 KT=8  -> 196608
    __bf16* Pp1 = Phh + 196608;       // N=256 K=256 KT=8  -> 65536
    __bf16* Pp2 = Pp1 + 65536;        // N=64  K=256 KT=8  -> 16384
    __bf16* Pq1 = Pp2 + 16384;        // N=256 K=512 KT=16 -> 131072
    __bf16* Pq2 = Pq1 + 131072;       // N=64  K=256 KT=8  -> 16384

    pack_kernel<<<16384  / 256, 256, 0, stream>>>(W_e1, Pe1,  64,  2, 16384);
    pack_kernel<<<65536  / 256, 256, 0, stream>>>(W_e2, Pe2, 256,  8, 65536);
    pack_kernel<<<49152  / 256, 256, 0, stream>>>(W_ih, Pih,  40,  2, 49152);
    pack_kernel<<<196608 / 256, 256, 0, stream>>>(W_hh, Phh, 256,  8, 196608);
    pack_kernel<<<65536  / 256, 256, 0, stream>>>(W_p1, Pp1, 256,  8, 65536);
    pack_kernel<<<16384  / 256, 256, 0, stream>>>(W_p2, Pp2, 256,  8, 16384);
    pack_kernel<<<131072 / 256, 256, 0, stream>>>(W_q1, Pq1, 512, 16, 131072);
    pack_kernel<<<16384  / 256, 256, 0, stream>>>(W_q2, Pq2, 256,  8, 16384);

    rssm_kernel<<<Bsz / 16, 256, 0, stream>>>(
        obs, actp, noi, b_e1, b_e2, b_ih, b_hh, b_p1, b_p2, b_q1, b_q2,
        Pe1, Pe2, Pih, Phh, Pp1, Pp2, Pq1, Pq2, out);
}
